// MultiHeadAttention_9861244912116
// MI455X (gfx1250) — matrix operand order repaired and checked by reading, not compiled, not run on hardware
//
#include <hip/hip_runtime.h>

// ---------------------------------------------------------------------------
// MHA forward for MI455X (gfx1250, wave32, WMMA).
// All matmuls via v_wmma_f32_16x16x32_f16 (f32 accumulate).
// Softmax runs in exp2 domain (log2e folded into Q scaling); softmax row sums
// are computed on the matrix pipe via a constant ones-column WMMA.
// ---------------------------------------------------------------------------

#define BDIM 4
#define SDIM 2048
#define DDIM 1024
#define NH   16
#define HD   64
#define NROWS (BDIM * SDIM)   // 8192
#define N_QKV (3 * DDIM)      // 3072

// (1/sqrt(64)) * log2(e): scores come out pre-scaled for exp2()
#define QSCALE_LOG2E 0.18033688011112042f

typedef __attribute__((ext_vector_type(16))) _Float16 v16h;
typedef __attribute__((ext_vector_type(8)))  _Float16 v8h;
typedef __attribute__((ext_vector_type(8)))  float    v8f;

#define WMMA_F16(a, b, c) \
  __builtin_amdgcn_wmma_f32_16x16x32_f16(false, (a), false, (b), (short)0, (c), false, false)

// A-fragment (16x32, M x K) from a row-major f16 matrix, leading dim `ld`
// (halves). ISA layout: lanes 0-15 -> K {0..7,16..23}; lanes 16-31 ->
// K {8..15,24..31}. Two contiguous 16-byte runs per lane -> b128 loads.
static __device__ __forceinline__ v16h load_a_frag(const _Float16* p0, int ld) {
  const int lane = threadIdx.x & 31;
  const _Float16* p = p0 + (lane & 15) * ld + ((lane >> 4) << 3);
  union { v16h v; v8h h[2]; } u;
  u.h[0] = *(const v8h*)(p);
  u.h[1] = *(const v8h*)(p + 16);
  return u.v;
}

// B-fragment (32x16, K x N) from an N-major f16 matrix (row = N, contiguous K),
// leading dim `ld`. ISA layout: lanes 0-15 -> K {0..7,16..23}; lanes 16-31 ->
// K {8..15,24..31}, N = lane%16. Two contiguous 16-byte runs -> two b128 loads.
static __device__ __forceinline__ v16h load_b_fragT(const _Float16* p0, int ld) {
  const int lane = threadIdx.x & 31;
  const _Float16* p = p0 + (lane & 15) * ld + ((lane >> 4) << 3);
  union { v16h v; v8h h[2]; } u;
  u.h[0] = *(const v8h*)(p);
  u.h[1] = *(const v8h*)(p + 16);
  return u.v;
}

// ---------------------------------------------------------------------------
// Conversion kernels (one-time, bandwidth-trivial)
// ---------------------------------------------------------------------------
__global__ void k_cvt(const float* __restrict__ in, _Float16* __restrict__ out, int n) {
  int i = blockIdx.x * blockDim.x + threadIdx.x;
  if (i < n) out[i] = (_Float16)in[i];
}

// in [K][N] row-major f32  ->  out [N][K] f16 (transposed)
__global__ void k_cvtT(const float* __restrict__ in, _Float16* __restrict__ out,
                       int K, int N) {
  int i = blockIdx.x * blockDim.x + threadIdx.x;
  if (i < K * N) {
    int k = i % K, nn = i / K;
    out[i] = (_Float16)in[k * N + nn];
  }
}

// ---------------------------------------------------------------------------
// QKV projection: [8192,1024]f16 @ [1024,3072] (+bias) with scatter epilogue.
// 8 waves/WG, each wave computes a 32x64 C tile (2x4 accumulators).
// Q scaled by log2e/sqrt(64); K stored [B,H,S,d]; V stored transposed [B,H,d,S].
// ---------------------------------------------------------------------------
__global__ __launch_bounds__(256) void k_gemm_qkv(
    const _Float16* __restrict__ xh,    // [8192,1024]
    const _Float16* __restrict__ wT,    // [3072,1024] (N-major)
    const float*    __restrict__ bias,  // [3072]
    _Float16* __restrict__ Qh,          // [B,H,S,d]
    _Float16* __restrict__ Kh,          // [B,H,S,d]
    _Float16* __restrict__ Vt)          // [B,H,d,S]
{
  const int w    = threadIdx.x >> 5;
  const int lane = threadIdx.x & 31;
  const int rowBase = blockIdx.y * 64  + (w & 1) * 32;
  const int colBase = blockIdx.x * 256 + (w >> 1) * 64;

  v8f acc[2][4] = {};
  for (int k0 = 0; k0 < DDIM; k0 += 32) {
    v16h a0 = load_a_frag(xh + rowBase * DDIM + k0, DDIM);
    v16h a1 = load_a_frag(xh + (rowBase + 16) * DDIM + k0, DDIM);
#pragma unroll
    for (int t = 0; t < 4; ++t) {
      v16h b = load_b_fragT(wT + (colBase + 16 * t) * DDIM + k0, DDIM);
      acc[0][t] = WMMA_F16(a0, b, acc[0][t]);
      acc[1][t] = WMMA_F16(a1, b, acc[1][t]);
    }
  }

  const int g  = lane >> 4, nn = lane & 15;
  const int part = colBase >> 10;          // 0=Q 1=K 2=V (wave-uniform)
  const int h    = (colBase >> 6) & 15;    // head (wave-uniform)
#pragma unroll
  for (int mi = 0; mi < 2; ++mi) {
#pragma unroll
    for (int t = 0; t < 4; ++t) {
      const int   cc = colBase + 16 * t + nn;
      const float bv = bias[cc];
      const int   dd = cc & 63;
#pragma unroll
      for (int r = 0; r < 8; ++r) {
        const int rr = rowBase + 16 * mi + r + 8 * g;
        const int bb = rr >> 11, ss = rr & 2047;
        const float val = acc[mi][t][r] + bv;
        const int idx = ((bb * NH + h) * SDIM + ss) * HD + dd;
        if (part == 0)      Qh[idx] = (_Float16)(val * QSCALE_LOG2E);
        else if (part == 1) Kh[idx] = (_Float16)val;
        else                Vt[((bb * NH + h) * HD + dd) * SDIM + ss] = (_Float16)val;
      }
    }
  }
}

// ---------------------------------------------------------------------------
// Flash attention: one wave owns 16 query rows, streams keys in chunks of 128.
// QK^T: 16 WMMAs/chunk; online softmax (exp2 domain) with shuffle row-max;
// row sums l come from a ones-column WMMA (matrix pipe, not VALU);
// P re-enters WMMA A-layout through a per-wave LDS tile; PV: 16 WMMAs/chunk.
// ---------------------------------------------------------------------------
__global__ __launch_bounds__(256) void k_attn(
    const _Float16* __restrict__ Qh,   // [B,H,S,d]
    const _Float16* __restrict__ Kh,   // [B,H,S,d]
    const _Float16* __restrict__ Vt,   // [B,H,d,S]
    _Float16* __restrict__ Oh)         // [B*S, 1024]
{
  __shared__ _Float16 lds_p[8 * 16 * 128];   // 4 KB per wave, private

  const int w    = threadIdx.x >> 5;
  const int lane = threadIdx.x & 31;
  const int g    = lane >> 4, nn = lane & 15;
  const int bh   = blockIdx.y;
  const int bb   = bh >> 4, h = bh & 15;
  const int q0   = blockIdx.x * 128 + w * 16;

  const _Float16* Qbase = Qh + (size_t)bh * SDIM * HD;
  const _Float16* Kbase = Kh + (size_t)bh * SDIM * HD;
  const _Float16* Vbase = Vt + (size_t)bh * HD * SDIM;
  _Float16* pbuf = lds_p + w * 16 * 128;

  v16h qa[2];
  qa[0] = load_a_frag(Qbase + q0 * HD, HD);
  qa[1] = load_a_frag(Qbase + q0 * HD + 32, HD);

  v16h ones;
#pragma unroll
  for (int e = 0; e < 16; ++e) ones[e] = (_Float16)1.0f;

  v8f o[4] = {};
  v8f lacc = {};                 // softmax row sums, accumulated by WMMA
  float m[8];
#pragma unroll
  for (int r = 0; r < 8; ++r) m[r] = -1e30f;

  for (int key0 = 0; key0 < SDIM; key0 += 128) {
    // ---- scores S = Q @ K^T for 16 q-rows x 128 keys (exp2-scaled) ----
    v8f sc[8] = {};
#pragma unroll
    for (int t = 0; t < 8; ++t) {
#pragma unroll
      for (int kc = 0; kc < 2; ++kc) {
        v16h b = load_b_fragT(Kbase + (key0 + 16 * t) * HD + 32 * kc, HD);
        sc[t] = WMMA_F16(qa[kc], b, sc[t]);
      }
    }

    // ---- per-row chunk max (reduce over 8 tiles, then over 16 lanes) ----
    v8f mx = sc[0];
#pragma unroll
    for (int t = 1; t < 8; ++t)
#pragma unroll
      for (int r = 0; r < 8; ++r) mx[r] = fmaxf(mx[r], sc[t][r]);
#pragma unroll
    for (int mask = 1; mask < 16; mask <<= 1)
#pragma unroll
      for (int r = 0; r < 8; ++r) mx[r] = fmaxf(mx[r], __shfl_xor(mx[r], mask, 32));

    float scale[8];
#pragma unroll
    for (int r = 0; r < 8; ++r) {
      const float mn = fmaxf(m[r], mx[r]);
      scale[r] = __builtin_amdgcn_exp2f(m[r] - mn);
      m[r] = mn;
    }

    // ---- rescale running output and row sums ----
#pragma unroll
    for (int t = 0; t < 4; ++t)
#pragma unroll
      for (int r = 0; r < 8; ++r) o[t][r] *= scale[r];
#pragma unroll
    for (int r = 0; r < 8; ++r) lacc[r] *= scale[r];

    // ---- P = exp2(S - m): C-layout -> LDS (row-major 16x128) ----
#pragma unroll
    for (int t = 0; t < 8; ++t)
#pragma unroll
      for (int r = 0; r < 8; ++r)
        pbuf[(r + 8 * g) * 128 + 16 * t + nn] =
            (_Float16)__builtin_amdgcn_exp2f(sc[t][r] - m[r]);
    __asm__ volatile("s_wait_dscnt 0" ::: "memory");  // same-wave LDS RAW

    // ---- O += P @ V ; l += P @ ones (rowsum on the matrix pipe) ----
#pragma unroll
    for (int kc = 0; kc < 4; ++kc) {
      v16h pa = load_a_frag(pbuf + 32 * kc, 128);
      lacc = WMMA_F16(pa, ones, lacc);
#pragma unroll
      for (int t = 0; t < 4; ++t) {
        v16h vb = load_b_fragT(Vbase + (16 * t) * SDIM + key0 + 32 * kc, SDIM);
        o[t] = WMMA_F16(pa, vb, o[t]);
      }
    }
  }

  // ---- normalize (rcp instead of 32 divides) and store to [B*S, H*d] ----
  float inv[8];
#pragma unroll
  for (int r = 0; r < 8; ++r) inv[r] = __builtin_amdgcn_rcpf(lacc[r]);
#pragma unroll
  for (int t = 0; t < 4; ++t)
#pragma unroll
    for (int r = 0; r < 8; ++r) {
      const int row = q0 + r + 8 * g;
      Oh[(size_t)(bb * SDIM + row) * DDIM + h * HD + 16 * t + nn] =
          (_Float16)(o[t][r] * inv[r]);
    }
}

// ---------------------------------------------------------------------------
// Output projection: [8192,1024]f16 @ [1024,1024] + bias -> f32 d_out
// ---------------------------------------------------------------------------
__global__ __launch_bounds__(256) void k_gemm_out(
    const _Float16* __restrict__ Oh,    // [8192,1024]
    const _Float16* __restrict__ wT,    // [1024,1024] (N-major)
    const float*    __restrict__ bias,  // [1024]
    float* __restrict__ out)            // [8192,1024]
{
  const int w    = threadIdx.x >> 5;
  const int lane = threadIdx.x & 31;
  const int rowBase = blockIdx.y * 64  + (w & 1) * 32;
  const int colBase = blockIdx.x * 256 + (w >> 1) * 64;

  v8f acc[2][4] = {};
  for (int k0 = 0; k0 < DDIM; k0 += 32) {
    v16h a0 = load_a_frag(Oh + rowBase * DDIM + k0, DDIM);
    v16h a1 = load_a_frag(Oh + (rowBase + 16) * DDIM + k0, DDIM);
#pragma unroll
    for (int t = 0; t < 4; ++t) {
      v16h b = load_b_fragT(wT + (colBase + 16 * t) * DDIM + k0, DDIM);
      acc[0][t] = WMMA_F16(a0, b, acc[0][t]);
      acc[1][t] = WMMA_F16(a1, b, acc[1][t]);
    }
  }

  const int g = lane >> 4, nn = lane & 15;
#pragma unroll
  for (int mi = 0; mi < 2; ++mi)
#pragma unroll
    for (int t = 0; t < 4; ++t) {
      const int   cc = colBase + 16 * t + nn;
      const float bv = bias[cc];
#pragma unroll
      for (int r = 0; r < 8; ++r) {
        const int rr = rowBase + 16 * mi + r + 8 * g;
        out[(size_t)rr * DDIM + cc] = acc[mi][t][r] + bv;
      }
    }
}

// ---------------------------------------------------------------------------
extern "C" void kernel_launch(void* const* d_in, const int* in_sizes, int n_in,
                              void* d_out, int out_size, void* d_ws, size_t ws_size,
                              hipStream_t stream) {
  const float* x      = (const float*)d_in[0];   // [4,2048,1024]
  const float* w_qkv  = (const float*)d_in[1];   // [1024,3072]
  const float* b_qkv  = (const float*)d_in[2];   // [3072]
  const float* w_out  = (const float*)d_in[3];   // [1024,1024]
  const float* b_out  = (const float*)d_in[4];   // [1024]
  float* out = (float*)d_out;

  // Workspace carve-up (all offsets multiples of 2 MB -> aligned). ~90 MB.
  char* ws = (char*)d_ws;
  _Float16* xh    = (_Float16*)ws;                       // 16 MB [8192,1024]
  _Float16* wqkvT = (_Float16*)(ws + (16u << 20));       //  6 MB [3072,1024]
  _Float16* woutT = (_Float16*)(ws + (22u << 20));       //  2 MB [1024,1024]
  _Float16* Qh    = (_Float16*)(ws + (24u << 20));       // 16 MB [B,H,S,d]
  _Float16* Kh    = (_Float16*)(ws + (40u << 20));       // 16 MB [B,H,S,d]
  _Float16* Vt    = (_Float16*)(ws + (56u << 20));       // 16 MB [B,H,d,S]
  _Float16* Oh    = (_Float16*)(ws + (72u << 20));       // 16 MB [8192,1024]

  const int nX = NROWS * DDIM;            // 8388608
  const int nWqkv = DDIM * N_QKV;         // 3145728
  const int nWout = DDIM * DDIM;          // 1048576

  k_cvt <<<(nX + 255) / 256, 256, 0, stream>>>(x, xh, nX);
  k_cvtT<<<(nWqkv + 255) / 256, 256, 0, stream>>>(w_qkv, wqkvT, DDIM, N_QKV);
  k_cvtT<<<(nWout + 255) / 256, 256, 0, stream>>>(w_out, woutT, DDIM, DDIM);

  k_gemm_qkv<<<dim3(N_QKV / 256, NROWS / 64), 256, 0, stream>>>(
      xh, wqkvT, b_qkv, Qh, Kh, Vt);

  k_attn<<<dim3(SDIM / 128, BDIM * NH), 256, 0, stream>>>(Qh, Kh, Vt, Oh);

  k_gemm_out<<<dim3(DDIM / 256, NROWS / 64), 256, 0, stream>>>(
      Oh, woutT, b_out, out);
}
